// MultiHeadAttention_52587579572839
// MI455X (gfx1250) — matrix operand order repaired and checked by reading, not compiled, not run on hardware
//
#include <hip/hip_runtime.h>

// ---------------- constants ----------------
constexpr int BB = 2;
constexpr int SS = 2048;
constexpr int DD = 1024;
constexpr int HH = 16;
constexpr int HDc = 64;              // head dim
constexpr float INV_SCALE = 0.125f;  // 1/sqrt(64)

// ---------------- types ----------------
typedef __attribute__((ext_vector_type(16))) __bf16        v16bf;
typedef __attribute__((ext_vector_type(8)))  float         v8f;
typedef __attribute__((ext_vector_type(4)))  float         v4f;
typedef __attribute__((ext_vector_type(4)))  unsigned int  v4u;

union FragB16 {            // 16 bf16 elements = 32 bytes
    v16bf          v;
    v4u            q[2];   // two 16-byte chunks
    unsigned short u[16];
};

static __device__ __forceinline__ __bf16 tobf(float f) { return (__bf16)f; }

static __device__ __forceinline__ v8f wmma_bf16(v16bf a, v16bf b, v8f c) {
    // D(f32 16x16) = A(bf16 16x32) * B(bf16 32x16) + C
    return __builtin_amdgcn_wmma_f32_16x16x32_bf16(
        /*neg_a=*/false, a, /*neg_b=*/false, b,
        /*c_mod=*/(short)0, c, /*reuse_a=*/false, /*reuse_b=*/false);
}

// A fragment (16x32 bf16), source bf16 row-major: row = base + lane%16,
// K chunks: [k0 + hi*8 .. +7] and [k0+16+hi*8 .. +7]  (ISA 16-bit A layout)
static __device__ __forceinline__ FragB16 loadA_bf16(const unsigned short* __restrict__ p,
                                                     int ld, int row, int k0,
                                                     int l16, int hi) {
    FragB16 a;
    const unsigned short* pa = p + (size_t)(row + l16) * ld + k0 + hi * 8;
    a.q[0] = *(const v4u*)(pa);
    a.q[1] = *(const v4u*)(pa + 16);
    return a;
}

// B fragment (32x16 bf16): lane = N column; K chunks [k0 + hi*8 .. +7] and
// [k0+16+hi*8 .. +7] from a bf16 buffer where K is the contiguous axis.
static __device__ __forceinline__ FragB16 loadB_bf16(const unsigned short* __restrict__ p,
                                                     int ld, int col, int k0,
                                                     int l16, int hi) {
    FragB16 b;
    const unsigned short* pb = p + (size_t)(col + l16) * ld + k0 + 8 * hi;
    b.q[0] = *(const v4u*)(pb);
    b.q[1] = *(const v4u*)(pb + 16);
    return b;
}

// B fragment from an LDS-staged 64x32 tile (row-major, ld = 32)
static __device__ __forceinline__ FragB16 loadB_lds(const unsigned short* sm,
                                                    int col, int l16, int hi) {
    FragB16 b;
    const unsigned short* pb = sm + (col + l16) * 32 + 8 * hi;
    b.q[0] = *(const v4u*)(pb);
    b.q[1] = *(const v4u*)(pb + 16);
    return b;
}

// Async global -> LDS copy of one 16-byte chunk (gfx1250, ASYNCcnt-tracked).
static __device__ __forceinline__ void async_g2l_b128(unsigned lds_addr,
                                                      const void* gptr) {
    unsigned long long ga = (unsigned long long)(size_t)gptr;
    asm volatile("global_load_async_to_lds_b128 %0, %1, off"
                 :: "v"(lds_addr), "v"(ga) : "memory");
}
static __device__ __forceinline__ void wait_async0() {
    asm volatile("s_wait_asynccnt 0x0" ::: "memory");
}

// ================= kernel 0: fp32 -> bf16 bulk convert =================
__global__ __launch_bounds__(256) void cvt_f32_bf16(const float* __restrict__ src,
                                                    unsigned short* __restrict__ dst,
                                                    int n4) {  // n/4 groups of 4
    const int i = blockIdx.x * 256 + threadIdx.x;
    if (i >= n4) return;
    const v4f f = *(const v4f*)(src + (size_t)i * 4);
    union { unsigned long long u64; unsigned short u[4]; } o;
#pragma unroll
    for (int j = 0; j < 4; ++j) o.u[j] = __builtin_bit_cast(unsigned short, tobf(f[j]));
    *(unsigned long long*)(dst + (size_t)i * 4) = o.u64;
}

// ================= kernel 1: fused QKV projection =================
// out[n,j] = x[n,:] . W[j,:] + bias[j]   (nn.Linear). z selects Q/K/V.
// Q,K stored bf16 as (b,h,s,hd); V stored bf16 transposed (b,h,hd,s).
// Wave tile: 32 rows x 64 cols. Shared W tile (64x32) is double-buffered in
// LDS via async global->LDS loads; 8 waves read it as ds_load_b128.
__global__ __launch_bounds__(256) void qkv_proj(
    const unsigned short* __restrict__ Xq, const unsigned short* __restrict__ Xk,
    const unsigned short* __restrict__ Xv,
    const unsigned short* __restrict__ Wq, const float* __restrict__ bq,
    const unsigned short* __restrict__ Wk, const float* __restrict__ bk,
    const unsigned short* __restrict__ Wv, const float* __restrict__ bv,
    unsigned short* __restrict__ Qb, unsigned short* __restrict__ Kb,
    unsigned short* __restrict__ Vtb) {
    __shared__ unsigned short smem[2][64 * 32];  // two 4KB W tiles

    const int proj = blockIdx.z;
    const unsigned short* X;
    const unsigned short* W;
    const float* bias;
    if (proj == 0)      { X = Xq; W = Wq; bias = bq; }
    else if (proj == 1) { X = Xk; W = Wk; bias = bk; }
    else                { X = Xv; W = Wv; bias = bv; }

    const int lane = threadIdx.x & 31;
    const int wave = threadIdx.x >> 5;
    const int l16 = lane & 15, hi = lane >> 4;
    const int row0 = blockIdx.y * 256 + wave * 32;  // n base (32 rows per wave)
    const int col0 = blockIdx.x * 64;               // j base (4 x 16 cols)

    const unsigned lds_base = (unsigned)(size_t)(void*)&smem[0][0];
    const int t = threadIdx.x;
    const int srow_t = t >> 2, schunk = t & 3;      // 64 rows x 4 chunks of 8 bf16

    auto stage = [&](int buf, int k0) {
        const unsigned short* g = W + (size_t)(col0 + srow_t) * DD + k0 + schunk * 8;
        const unsigned lds = lds_base + (unsigned)(buf * 4096 + srow_t * 64 + schunk * 16);
        async_g2l_b128(lds, g);
    };

    stage(0, 0);
    wait_async0();
    __syncthreads();

    v8f acc[2][4] = {};
    int buf = 0;
    for (int k0 = 0; k0 < DD; k0 += 32) {
        if (k0 + 32 < DD) stage(buf ^ 1, k0 + 32);

        FragB16 a0 = loadA_bf16(X, DD, row0, k0, l16, hi);
        FragB16 a1 = loadA_bf16(X, DD, row0 + 16, k0, l16, hi);
        if (k0 + 32 < DD)
            __builtin_prefetch(X + (size_t)(row0 + l16) * DD + k0 + 32, 0, 3);

        const unsigned short* sm = &smem[buf][0];
#pragma unroll
        for (int s = 0; s < 4; ++s) {
            FragB16 b = loadB_lds(sm, s * 16, l16, hi);
            acc[0][s] = wmma_bf16(a0.v, b.v, acc[0][s]);
            acc[1][s] = wmma_bf16(a1.v, b.v, acc[1][s]);
        }
        wait_async0();
        __syncthreads();
        buf ^= 1;
    }

    // store: C/D layout -> M = r + 8*hi, N = l16
#pragma unroll
    for (int tt = 0; tt < 2; ++tt)
#pragma unroll
        for (int s = 0; s < 4; ++s) {
            const int j = col0 + s * 16 + l16;
            const float bj = bias[j];
            const int h = j >> 6, hd = j & 63;
#pragma unroll
            for (int r = 0; r < 8; ++r) {
                const int n = row0 + tt * 16 + r + 8 * hi;
                const int b = n >> 11, srow = n & (SS - 1);
                const unsigned short o =
                    __builtin_bit_cast(unsigned short, tobf(acc[tt][s][r] + bj));
                const size_t bh = (size_t)(b * HH + h);
                if (proj == 0)      Qb[(bh * SS + srow) * HDc + hd] = o;
                else if (proj == 1) Kb[(bh * SS + srow) * HDc + hd] = o;
                else                Vtb[(bh * HDc + hd) * SS + srow] = o;
            }
        }
}

// ================= kernel 2: scores = Q.K^T / sqrt(hd), causal mask ======
// wave: 16 q-rows x 64 k-cols (4 accumulators, A reused). Writes raw scores
// (or -inf above the diagonal) into the attn_weights region of d_out.
__global__ __launch_bounds__(256) void scores_kernel(
    const unsigned short* __restrict__ Qb, const unsigned short* __restrict__ Kb,
    float* __restrict__ attn) {
    const int bh = blockIdx.z;
    const int q0 = blockIdx.y * 16;
    const int wave = threadIdx.x >> 5;
    const int c0 = blockIdx.x * 512 + wave * 64;
    const int lane = threadIdx.x & 31, l16 = lane & 15, hi = lane >> 4;

    float* out = attn + ((size_t)bh * SS + q0) * SS;

    if (c0 > q0 + 15) {  // tile entirely above diagonal: masked
#pragma unroll
        for (int s = 0; s < 4; ++s)
#pragma unroll
            for (int r = 0; r < 8; ++r)
                out[(size_t)(r + 8 * hi) * SS + (c0 + s * 16 + l16)] = -__builtin_inff();
        return;
    }

    const unsigned short* Qp = Qb + ((size_t)bh * SS + q0) * HDc;
    const unsigned short* Kp = Kb + (size_t)bh * SS * HDc;

    v8f acc[4] = {};
#pragma unroll
    for (int k0 = 0; k0 < HDc; k0 += 32) {
        FragB16 a;
        const unsigned short* pa = Qp + (size_t)l16 * HDc + k0 + hi * 8;
        a.q[0] = *(const v4u*)(pa);
        a.q[1] = *(const v4u*)(pa + 16);
#pragma unroll
        for (int s = 0; s < 4; ++s) {
            FragB16 b = loadB_bf16(Kp, HDc, c0 + s * 16, k0, l16, hi);
            acc[s] = wmma_bf16(a.v, b.v, acc[s]);
        }
    }
#pragma unroll
    for (int s = 0; s < 4; ++s)
#pragma unroll
        for (int r = 0; r < 8; ++r) {
            const int q = q0 + r + 8 * hi;
            const int k = c0 + s * 16 + l16;
            const float v = acc[s][r] * INV_SCALE;
            out[(size_t)(r + 8 * hi) * SS + k] = (k > q) ? -__builtin_inff() : v;
        }
}

// ================= kernel 3: row softmax, in place =================
__global__ __launch_bounds__(256) void softmax_kernel(float* __restrict__ attn) {
    __shared__ float red[256];
    float* p = attn + (size_t)blockIdx.x * SS;
    const int t = threadIdx.x;

    float m = -__builtin_inff();
    for (int i = t; i < SS; i += 256) m = fmaxf(m, p[i]);
    red[t] = m; __syncthreads();
    for (int s = 128; s > 0; s >>= 1) { if (t < s) red[t] = fmaxf(red[t], red[t + s]); __syncthreads(); }
    m = red[0]; __syncthreads();

    float sum = 0.f;
    for (int i = t; i < SS; i += 256) { float e = __expf(p[i] - m); p[i] = e; sum += e; }
    red[t] = sum; __syncthreads();
    for (int s = 128; s > 0; s >>= 1) { if (t < s) red[t] += red[t + s]; __syncthreads(); }
    const float inv = 1.0f / red[0];
    for (int i = t; i < SS; i += 256) p[i] *= inv;
}

// ================= kernel 4: attn_out = P . V (bf16 out) =================
// wave: 16 q-rows x 64 hd-cols (A reused over 4 B frags); causal K truncation.
__global__ __launch_bounds__(256) void pv_kernel(
    const float* __restrict__ attn, const unsigned short* __restrict__ Vtb,
    unsigned short* __restrict__ AOb) {
    const int bh = blockIdx.z;
    const int wave = threadIdx.x >> 5;
    const int q0 = blockIdx.y * 128 + wave * 16;
    const int lane = threadIdx.x & 31, l16 = lane & 15, hi = lane >> 4;

    const float* P = attn + ((size_t)bh * SS + q0) * SS;
    const unsigned short* Vp = Vtb + (size_t)bh * HDc * SS;

    v8f acc[4] = {};
    const int kmax = q0 + 16;  // weights are zero for k > q0+15
    for (int k0 = 0; k0 < kmax; k0 += 32) {
        FragB16 a;
        const float* pa = P + (size_t)l16 * SS + k0 + hi * 8;
#pragma unroll
        for (int i = 0; i < 8; ++i) a.v[i] = tobf(pa[i]);
#pragma unroll
        for (int i = 0; i < 8; ++i) a.v[8 + i] = tobf(pa[16 + i]);
        if (k0 + 32 < kmax) __builtin_prefetch(pa + 32, 0, 3);
#pragma unroll
        for (int s = 0; s < 4; ++s) {
            FragB16 b = loadB_bf16(Vp, SS, s * 16, k0, l16, hi);
            acc[s] = wmma_bf16(a.v, b.v, acc[s]);
        }
    }
    const int b = bh >> 4, h = bh & 15;
#pragma unroll
    for (int s = 0; s < 4; ++s)
#pragma unroll
        for (int r = 0; r < 8; ++r) {
            const int srow = q0 + r + 8 * hi;
            const int col = h * HDc + s * 16 + l16;
            AOb[(size_t)(b * SS + srow) * DD + col] =
                __builtin_bit_cast(unsigned short, tobf(acc[s][r]));
        }
}

// ================= kernel 5: output projection =================
// Wave tile 32x64; Wo tile double-buffered in LDS via async loads; fp32 out.
__global__ __launch_bounds__(256) void oproj_kernel(
    const unsigned short* __restrict__ AOb, const unsigned short* __restrict__ Wob,
    const float* __restrict__ bo, float* __restrict__ out) {
    __shared__ unsigned short smem[2][64 * 32];

    const int lane = threadIdx.x & 31;
    const int wave = threadIdx.x >> 5;
    const int l16 = lane & 15, hi = lane >> 4;
    const int row0 = blockIdx.y * 256 + wave * 32;
    const int col0 = blockIdx.x * 64;

    const unsigned lds_base = (unsigned)(size_t)(void*)&smem[0][0];
    const int t = threadIdx.x;
    const int srow_t = t >> 2, schunk = t & 3;

    auto stage = [&](int buf, int k0) {
        const unsigned short* g = Wob + (size_t)(col0 + srow_t) * DD + k0 + schunk * 8;
        const unsigned lds = lds_base + (unsigned)(buf * 4096 + srow_t * 64 + schunk * 16);
        async_g2l_b128(lds, g);
    };

    stage(0, 0);
    wait_async0();
    __syncthreads();

    v8f acc[2][4] = {};
    int buf = 0;
    for (int k0 = 0; k0 < DD; k0 += 32) {
        if (k0 + 32 < DD) stage(buf ^ 1, k0 + 32);

        FragB16 a0 = loadA_bf16(AOb, DD, row0, k0, l16, hi);
        FragB16 a1 = loadA_bf16(AOb, DD, row0 + 16, k0, l16, hi);
        if (k0 + 32 < DD)
            __builtin_prefetch(AOb + (size_t)(row0 + l16) * DD + k0 + 32, 0, 3);

        const unsigned short* sm = &smem[buf][0];
#pragma unroll
        for (int s = 0; s < 4; ++s) {
            FragB16 b = loadB_lds(sm, s * 16, l16, hi);
            acc[0][s] = wmma_bf16(a0.v, b.v, acc[0][s]);
            acc[1][s] = wmma_bf16(a1.v, b.v, acc[1][s]);
        }
        wait_async0();
        __syncthreads();
        buf ^= 1;
    }
#pragma unroll
    for (int tt = 0; tt < 2; ++tt)
#pragma unroll
        for (int s = 0; s < 4; ++s) {
            const int j = col0 + s * 16 + l16;
            const float bj = bo[j];
#pragma unroll
            for (int r = 0; r < 8; ++r) {
                const int n = row0 + tt * 16 + r + 8 * hi;
                out[(size_t)n * DD + j] = acc[tt][s][r] + bj;
            }
        }
}

// ================= launch =================
extern "C" void kernel_launch(void* const* d_in, const int* in_sizes, int n_in,
                              void* d_out, int out_size, void* d_ws, size_t ws_size,
                              hipStream_t stream) {
    const float* query = (const float*)d_in[0];
    const float* key_t = (const float*)d_in[1];
    const float* value = (const float*)d_in[2];
    // d_in[3] = bool causal mask, recomputed analytically
    const float* Wq = (const float*)d_in[4];
    const float* bq = (const float*)d_in[5];
    const float* Wk = (const float*)d_in[6];
    const float* bk = (const float*)d_in[7];
    const float* Wv = (const float*)d_in[8];
    const float* bv = (const float*)d_in[9];
    const float* Wo = (const float*)d_in[10];
    const float* bo = (const float*)d_in[11];

    float* out  = (float*)d_out;                       // (B,S,D)
    float* attn = out + (size_t)BB * SS * DD;          // (B,H,S,S)

    const size_t E = (size_t)BB * SS * DD;             // B*S*D elements
    const size_t M = (size_t)DD * DD;                  // D*D elements
    unsigned short* p   = (unsigned short*)d_ws;
    unsigned short* Qb  = p;  p += E;                  // bf16 (b,h,s,hd)
    unsigned short* Kb  = p;  p += E;                  // bf16 (b,h,s,hd)
    unsigned short* Vtb = p;  p += E;                  // bf16 (b,h,hd,s)
    unsigned short* AOb = p;  p += E;                  // bf16 (b,s,D)
    unsigned short* Xqb = p;  p += E;                  // bf16 inputs
    unsigned short* Xkb = p;  p += E;
    unsigned short* Xvb = p;  p += E;
    unsigned short* Wqb = p;  p += M;                  // bf16 weights
    unsigned short* Wkb = p;  p += M;
    unsigned short* Wvb = p;  p += M;
    unsigned short* Wob = p;  p += M;

    // ---- bulk fp32 -> bf16 conversion (hoisted out of GEMM inner loops) ----
    const int E4 = (int)(E / 4), M4 = (int)(M / 4);
    cvt_f32_bf16<<<dim3((E4 + 255) / 256), 256, 0, stream>>>(query, Xqb, E4);
    cvt_f32_bf16<<<dim3((E4 + 255) / 256), 256, 0, stream>>>(key_t, Xkb, E4);
    cvt_f32_bf16<<<dim3((E4 + 255) / 256), 256, 0, stream>>>(value, Xvb, E4);
    cvt_f32_bf16<<<dim3((M4 + 255) / 256), 256, 0, stream>>>(Wq, Wqb, M4);
    cvt_f32_bf16<<<dim3((M4 + 255) / 256), 256, 0, stream>>>(Wk, Wkb, M4);
    cvt_f32_bf16<<<dim3((M4 + 255) / 256), 256, 0, stream>>>(Wv, Wvb, M4);
    cvt_f32_bf16<<<dim3((M4 + 255) / 256), 256, 0, stream>>>(Wo, Wob, M4);

    qkv_proj<<<dim3(DD / 64, (BB * SS) / 256, 3), 256, 0, stream>>>(
        Xqb, Xkb, Xvb, Wqb, bq, Wkb, bk, Wvb, bv, Qb, Kb, Vtb);

    scores_kernel<<<dim3(SS / 512, SS / 16, BB * HH), 256, 0, stream>>>(Qb, Kb, attn);

    softmax_kernel<<<dim3(BB * HH * SS), 256, 0, stream>>>(attn);

    pv_kernel<<<dim3(1, SS / 128, BB * HH), 256, 0, stream>>>(attn, Vtb, AOb);

    oproj_kernel<<<dim3(DD / 64, (BB * SS) / 256, 1), 256, 0, stream>>>(AOb, Wob, bo, out);
}
